// SimpleRNN_25709674234260
// MI455X (gfx1250) — matrix operand order repaired and checked by reading, not compiled, not run on hardware
//
#include <hip/hip_runtime.h>

// ---------------------------------------------------------------------------
// Problem dimensions (fixed by the reference)
// ---------------------------------------------------------------------------
#define S_LEN  8192
#define E_DIM  1024
#define H_DIM  2048
#define T_TAGS 50
#define T_PAD  64            // tags padded to 4 WMMA N-tiles

#define RNN_NWG   64                  // persistent workgroups for the recurrence
#define RNN_ROWS  (H_DIM / RNN_NWG)   // 32 rows of W_hh per WGP
#define W_PITCH   2052                // LDS row pitch (dwords): 16B-aligned, de-conflicted

// gfx1250 async global->LDS copy path (ASYNCcnt-tracked), if the toolchain has it
#if defined(__HIP_DEVICE_COMPILE__) && defined(__has_builtin)
#if __has_builtin(__builtin_amdgcn_global_load_async_to_lds_b128) && \
    __has_builtin(__builtin_amdgcn_s_wait_asynccnt)
#define HAVE_ASYNC_LDS 1
#endif
#endif

// ---------------------------------------------------------------------------
// WMMA vector types (CDNA5 gfx1250, wave32)
// ---------------------------------------------------------------------------
typedef __attribute__((ext_vector_type(16))) __bf16 v16bf;
typedef __attribute__((ext_vector_type(8)))  __bf16 v8bf;
typedef __attribute__((ext_vector_type(8)))  float  v8f;

union V16U { v16bf v; v8bf h[2]; };

// round-to-nearest-even f32 -> bf16
static __device__ __forceinline__ unsigned short f2bf(float f) {
  union { float f; unsigned u; } x; x.f = f;
  unsigned r = x.u + 0x7FFFu + ((x.u >> 16) & 1u);
  return (unsigned short)(r >> 16);
}

// ---------------------------------------------------------------------------
// Kernel 1: embedding gather + cast to bf16
// ---------------------------------------------------------------------------
__global__ void k_embed_cast(const int* __restrict__ sent,
                             const float* __restrict__ emb,
                             unsigned short* __restrict__ xbf) {
  const int base = (blockIdx.x * 256 + threadIdx.x) * 4;
  const int s = base >> 10;            // / E_DIM
  const int e = base & (E_DIM - 1);
  const float4 v = *(const float4*)(emb + (size_t)sent[s] * E_DIM + e);
  ushort4 o;
  o.x = f2bf(v.x); o.y = f2bf(v.y); o.z = f2bf(v.z); o.w = f2bf(v.w);
  *(ushort4*)(xbf + base) = o;
}

// ---------------------------------------------------------------------------
// Kernel 2: generic f32 -> bf16 cast (for W_ih)
// ---------------------------------------------------------------------------
__global__ void k_cast_bf16(const float* __restrict__ src,
                            unsigned short* __restrict__ dst) {
  const int base = (blockIdx.x * 256 + threadIdx.x) * 4;
  const float4 v = *(const float4*)(src + base);
  ushort4 o;
  o.x = f2bf(v.x); o.y = f2bf(v.y); o.z = f2bf(v.z); o.w = f2bf(v.w);
  *(ushort4*)(dst + base) = o;
}

// ---------------------------------------------------------------------------
// Kernel 3: W_out [50][2048] -> bf16 padded to [64][2048]
// ---------------------------------------------------------------------------
__global__ void k_cast_wout(const float* __restrict__ wout,
                            unsigned short* __restrict__ dst) {
  const int base = (blockIdx.x * 256 + threadIdx.x) * 4;
  const int row = base >> 11;          // / H_DIM
  const int col = base & (H_DIM - 1);
  ushort4 o;
  if (row < T_TAGS) {
    const float4 v = *(const float4*)(wout + row * H_DIM + col);
    o.x = f2bf(v.x); o.y = f2bf(v.y); o.z = f2bf(v.z); o.w = f2bf(v.w);
  } else {
    o.x = 0; o.y = 0; o.z = 0; o.w = 0;
  }
  *(ushort4*)(dst + base) = o;
}

// ---------------------------------------------------------------------------
// Kernel 4: xi = x_bf16 @ W_ih_bf16^T + (b_ih + b_hh)
// Register-blocked: each wave computes a 64x32 block = 4M x 2N WMMA tiles.
// Per K-step: 8 A-loads + 4 B-loads (b128-equiv 12) feed 8 v_wmma ops.
// A (16x32 bf16): lane l: M = l&15, K = (l>>4)*16 .. +15
// B (32x16 bf16): lane l: N = l&15, K = (l>>4)*16 .. +15  (contiguous in W row)
// C (16x16 f32) : VGPR r: M = r + 8*(l>>4), N = l&15
// ---------------------------------------------------------------------------
__global__ void k_gemm_xi(const unsigned short* __restrict__ X,   // [S][E] bf16
                          const unsigned short* __restrict__ W,   // [H][E] bf16
                          const float* __restrict__ b_ih,
                          const float* __restrict__ b_hh,
                          float* __restrict__ xi) {               // [S][H] f32
  const int lane = threadIdx.x & 31;
  const int wave = threadIdx.x >> 5;
  const int job  = blockIdx.x * 8 + wave;   // (512/4)*(128/2) = 8192 jobs
  const int NG   = (H_DIM / 16) / 2;        // 64 n-groups
  const int mg = job / NG, ng = job % NG;   // mg: 0..127, ng: 0..63
  const int l16 = lane & 15, lhi = lane >> 4;
  const int kb  = lhi * 8;

  const unsigned short* pa[4];
  const unsigned short* pb[2];
#pragma unroll
  for (int mt = 0; mt < 4; ++mt)
    pa[mt] = X + (size_t)(mg * 64 + mt * 16 + l16) * E_DIM;
#pragma unroll
  for (int nt = 0; nt < 2; ++nt)
    pb[nt] = W + (size_t)(ng * 32 + nt * 16 + l16) * E_DIM + lhi * 16;

  v8f acc[4][2] = {};
  for (int kk = 0; kk < E_DIM; kk += 32) {
    V16U a[4];
    v16bf b[2];
#pragma unroll
    for (int mt = 0; mt < 4; ++mt) {
      a[mt].h[0] = *(const v8bf*)(pa[mt] + kk + lhi * 16);
      a[mt].h[1] = *(const v8bf*)(pa[mt] + kk + lhi * 16 + 8);
    }
#pragma unroll
    for (int nt = 0; nt < 2; ++nt)
      b[nt] = *(const v16bf*)(pb[nt] + kk);
#pragma unroll
    for (int mt = 0; mt < 4; ++mt)
#pragma unroll
      for (int nt = 0; nt < 2; ++nt)
        acc[mt][nt] = __builtin_amdgcn_wmma_f32_16x16x32_bf16(
            false, a[mt].v, false, b[nt], (short)0, acc[mt][nt], false, false);
  }

#pragma unroll
  for (int nt = 0; nt < 2; ++nt) {
    const int   n    = ng * 32 + nt * 16 + l16;
    const float bias = b_ih[n] + b_hh[n];
#pragma unroll
    for (int mt = 0; mt < 4; ++mt) {
      float* po = xi + (size_t)(mg * 64 + mt * 16 + lhi * 8) * H_DIM + n;
#pragma unroll
      for (int r = 0; r < 8; ++r) po[(size_t)r * H_DIM] = acc[mt][nt][r] + bias;
    }
  }
}

// ---------------------------------------------------------------------------
// Grid-wide generation barrier (persistent cooperative kernel)
// ---------------------------------------------------------------------------
static __device__ __forceinline__ void grid_barrier(unsigned* bar, int nwg, int t) {
  __syncthreads();
  if (threadIdx.x == 0) {
    __threadfence();
    const unsigned prev = atomicAdd(&bar[0], 1u);
    if (prev == (unsigned)(nwg - 1)) {
      bar[0] = 0;
      __threadfence();
      atomicAdd(&bar[1], 1u);           // release generation t+1
    } else {
      while (__hip_atomic_load(&bar[1], __ATOMIC_ACQUIRE,
                               __HIP_MEMORY_SCOPE_AGENT) < (unsigned)(t + 1)) {
        __builtin_amdgcn_s_sleep(8);
      }
    }
  }
  __syncthreads();
}

// ---------------------------------------------------------------------------
// Kernel 5: Elman recurrence, persistent across all 8192 steps.
// 64 WGs x 256 threads; each WG pins its 32x2048 fp32 W_hh slice in LDS at
// pitch 2052 dwords (lane r -> bank group 4r+c mod 64: conflict-minimized
// ds_load_b128). Mapping: lane = row (32 rows), wave = 256-col segment (8).
// h segment reads are same-address broadcasts. Per-step h staging uses the
// gfx1250 async global->LDS copy when available (ASYNCcnt).
// ---------------------------------------------------------------------------
__global__ void k_recurrence(const float* __restrict__ Whh,      // [H][H] f32
                             const float* __restrict__ xi,       // [S][H] f32
                             float* __restrict__ hbuf,           // [2][H] f32
                             unsigned short* __restrict__ rnn,   // [S][H] bf16
                             unsigned* __restrict__ bar) {
  extern __shared__ float smem[];
  float* sW   = smem;                                   // 32 * W_PITCH
  float* sh   = smem + RNN_ROWS * W_PITCH;              // H_DIM
  float* part = smem + RNN_ROWS * W_PITCH + H_DIM;      // 8 * 32

  const int tid  = threadIdx.x;
  const int row0 = blockIdx.x * RNN_ROWS;

  // Pin W_hh slice into LDS once (row pitch W_PITCH for bank spread).
  for (int i = tid; i < RNN_ROWS * (H_DIM / 4); i += 256) {
    const int rr = i >> 9;             // / 512 float4-chunks per row
    const int c4 = (i & 511) * 4;
    const float4 v = *(const float4*)(Whh + (size_t)(row0 + rr) * H_DIM + c4);
    *(float4*)(sW + rr * W_PITCH + c4) = v;
  }
  __syncthreads();

  const int lane = tid & 31;          // local row 0..31
  const int wv   = tid >> 5;          // column segment 0..7
  const float4* wr4 = (const float4*)(sW + lane * W_PITCH + wv * 256);

  for (int t = 0; t < S_LEN; ++t) {
    // ---- stage h_t (8 KB) global -> LDS ----
    const float* hsrc = hbuf + (t & 1) * H_DIM;
#if defined(HAVE_ASYNC_LDS)
    {
      typedef int v4i_t __attribute__((ext_vector_type(4)));
      typedef __attribute__((address_space(1))) v4i_t as1_v4i;  // global
      typedef __attribute__((address_space(3))) v4i_t as3_v4i;  // LDS
      as1_v4i* gp  = (as1_v4i*)((char*)hsrc + tid * 16);
      as3_v4i* lp  = (as3_v4i*)((char*)sh + tid * 16);
      __builtin_amdgcn_global_load_async_to_lds_b128(gp, lp, 0, 0);
      as1_v4i* gp2 = (as1_v4i*)((char*)hsrc + 4096 + tid * 16);
      as3_v4i* lp2 = (as3_v4i*)((char*)sh + 4096 + tid * 16);
      __builtin_amdgcn_global_load_async_to_lds_b128(gp2, lp2, 0, 0);
      __builtin_amdgcn_s_wait_asynccnt(0);
    }
#else
    {
      const float4* s4 = (const float4*)hsrc;
      float4*       d4 = (float4*)sh;
      for (int i = tid; i < H_DIM / 4; i += 256) d4[i] = s4[i];
    }
#endif
    __syncthreads();

    // ---- 256-wide fp32 dot segment from LDS (b128 reads) ----
    const float4* hv4 = (const float4*)(sh + wv * 256);
    float acc = 0.f;
#pragma unroll 8
    for (int i = 0; i < 64; ++i) {
      const float4 a = wr4[i];
      const float4 b = hv4[i];
      acc = fmaf(a.x, b.x, fmaf(a.y, b.y, fmaf(a.z, b.z, fmaf(a.w, b.w, acc))));
    }
    part[wv * 32 + lane] = acc;
    __syncthreads();

    // ---- reduce 8 segments, tanh, publish ----
    if (tid < 32) {
      float s = 0.f;
#pragma unroll
      for (int w = 0; w < 8; ++w) s += part[w * 32 + tid];
      const int  grow = row0 + tid;
      const float hn  = tanhf(xi[(size_t)t * H_DIM + grow] + s);
      hbuf[((t + 1) & 1) * H_DIM + grow] = hn;
      rnn[(size_t)t * H_DIM + grow] = f2bf(hn);
    }
    grid_barrier(bar, RNN_NWG, t);
  }
}

// ---------------------------------------------------------------------------
// Kernel 6: tag head (bf16 WMMA) + log_softmax over 50 tags.
// 256 threads = 8 waves; wave = one 16-row M-tile computing ALL 4 N-tiles
// with a shared A-fragment (A loaded once per K-step).
// ---------------------------------------------------------------------------
__global__ void k_head(const unsigned short* __restrict__ R,    // [S][H] bf16
                       const unsigned short* __restrict__ WO,   // [64][H] bf16
                       const float* __restrict__ b_out,         // [50]
                       float* __restrict__ out) {               // [S][50]
  __shared__ float tag[8][16 * T_PAD];                          // 32 KB
  const int lane = threadIdx.x & 31;
  const int wave = threadIdx.x >> 5;          // M-tile within block
  const int tm   = blockIdx.x * 8 + wave;     // 0..4095
  const int l16 = lane & 15, lhi = lane >> 4;
  const int kb  = lhi * 8;

  const unsigned short* pa = R + (size_t)(tm * 16 + l16) * H_DIM;
  const unsigned short* pb[4];
#pragma unroll
  for (int nt = 0; nt < 4; ++nt)
    pb[nt] = WO + (size_t)(nt * 16 + l16) * H_DIM + lhi * 16;

  v8f acc[4] = {};
  for (int kk = 0; kk < H_DIM; kk += 32) {
    V16U a;
    a.h[0] = *(const v8bf*)(pa + kk + lhi * 16);
    a.h[1] = *(const v8bf*)(pa + kk + lhi * 16 + 8);
#pragma unroll
    for (int nt = 0; nt < 4; ++nt) {
      const v16bf b = *(const v16bf*)(pb[nt] + kk);
      acc[nt] = __builtin_amdgcn_wmma_f32_16x16x32_bf16(
          false, a.v, false, b, (short)0, acc[nt], false, false);
    }
  }
#pragma unroll
  for (int nt = 0; nt < 4; ++nt) {
    const int   col  = nt * 16 + l16;
    const float bias = (col < T_TAGS) ? b_out[col] : 0.f;
#pragma unroll
    for (int r = 0; r < 8; ++r)
      tag[wave][(lhi * 8 + r) * T_PAD + col] = acc[nt][r] + bias;
  }
  __syncthreads();

  if (threadIdx.x < 128) {                    // 8 tiles x 16 rows
    const int mt  = threadIdx.x >> 4;
    const int row = threadIdx.x & 15;
    const float* tr = tag[mt] + row * T_PAD;
    float m = tr[0];
    for (int c = 1; c < T_TAGS; ++c) m = fmaxf(m, tr[c]);
    float ssum = 0.f;
    for (int c = 0; c < T_TAGS; ++c) ssum += __expf(tr[c] - m);
    const float lse = m + __logf(ssum);
    float* po = out + (size_t)((blockIdx.x * 8 + mt) * 16 + row) * T_TAGS;
    for (int c = 0; c < T_TAGS; ++c) po[c] = tr[c] - lse;
  }
}

// ---------------------------------------------------------------------------
// Host orchestration (graph-capture safe: only stream ops, deterministic)
// ---------------------------------------------------------------------------
extern "C" void kernel_launch(void* const* d_in, const int* in_sizes, int n_in,
                              void* d_out, int out_size, void* d_ws, size_t ws_size,
                              hipStream_t stream) {
  const int*   sentence = (const int*)  d_in[0];
  const float* emb      = (const float*)d_in[1];
  const float* W_ih     = (const float*)d_in[2];
  const float* W_hh     = (const float*)d_in[3];
  const float* b_ih     = (const float*)d_in[4];
  const float* b_hh     = (const float*)d_in[5];
  const float* W_out    = (const float*)d_in[6];
  const float* b_out    = (const float*)d_in[7];
  const float* h0       = (const float*)d_in[8];

  char*  ws  = (char*)d_ws;
  size_t off = 0;
  auto alloc = [&](size_t bytes) -> void* {
    void* p = ws + off;
    off = (off + bytes + 255) & ~(size_t)255;
    return p;
  };
  unsigned short* xbf  = (unsigned short*)alloc((size_t)S_LEN * E_DIM * 2);
  unsigned short* wbf  = (unsigned short*)alloc((size_t)H_DIM * E_DIM * 2);
  float*          xi   = (float*)         alloc((size_t)S_LEN * H_DIM * 4);
  unsigned short* rnn  = (unsigned short*)alloc((size_t)S_LEN * H_DIM * 2);
  unsigned short* wout = (unsigned short*)alloc((size_t)T_PAD * H_DIM * 2);
  float*          hbuf = (float*)         alloc((size_t)2 * H_DIM * 4);
  unsigned*       bar  = (unsigned*)      alloc(256);

  (void)hipMemsetAsync(bar, 0, 256, stream);
  (void)hipMemcpyAsync(hbuf, h0, H_DIM * sizeof(float),
                       hipMemcpyDeviceToDevice, stream);

  k_embed_cast<<<S_LEN * E_DIM / 1024, 256, 0, stream>>>(sentence, emb, xbf);
  k_cast_bf16 <<<H_DIM * E_DIM / 1024, 256, 0, stream>>>(W_ih, wbf);
  k_cast_wout <<<T_PAD * H_DIM / 1024, 256, 0, stream>>>(W_out, wout);

  // 8192 wave-jobs (4Mx2N tiles each), 8 waves/block
  k_gemm_xi<<<1024, 256, 0, stream>>>(xbf, wbf, b_ih, b_hh, xi);

  const size_t ldsBytes =
      (size_t)(RNN_ROWS * W_PITCH + H_DIM + 8 * 32) * sizeof(float);
  k_recurrence<<<RNN_NWG, 256, ldsBytes, stream>>>(W_hh, xi, hbuf, rnn, bar);

  k_head<<<S_LEN / 16, 256, 0, stream>>>(rnn, wout, b_out, (float*)d_out);
}
